// Attention2_84679575208597
// MI455X (gfx1250) — matrix operand order repaired and checked by reading, not compiled, not run on hardware
//
#include <hip/hip_runtime.h>
#include <hip/hip_bf16.h>
#include <stdint.h>

// ---------------------------------------------------------------------------
// CDNA5 (gfx1250) attention pipeline.
// All GEMMs via v_wmma_f32_16x16x32_bf16, double-buffered LDS staging,
// async global->LDS copies (ASYNCcnt) for bf16-sourced tiles,
// v_perm_b32-based fp32->bf16 packing on paired paths; B[K][N] staging and Cb use scalar cvt.
// ---------------------------------------------------------------------------

typedef __attribute__((ext_vector_type(16))) __bf16 v16bf;
typedef __attribute__((ext_vector_type(8)))  float  v8f;

namespace cfg {
constexpr int NB   = 16;    // batch
constexpr int KL   = 2048;  // key length
constexpr int QL   = 512;   // query length
constexpr int EMB  = 1024;
constexpr int HID  = 1024;

constexpr int BM = 128, BN = 128, BK = 32;  // block tile
constexpr int LDSS = 40;                    // padded bf16 LDS row stride
constexpr int TILE = 128 * LDSS;            // elems per LDS tile buffer
}

template <typename T> struct IsBf16 { static constexpr bool value = false; };
template <> struct IsBf16<unsigned short> { static constexpr bool value = true; };

__device__ __forceinline__ unsigned f32bits(float f) {
  union { float f; unsigned u; } x; x.f = f;
  return x.u;
}
// round-half-up bf16 (differs from RNE only on exact ties)
__device__ __forceinline__ unsigned short cvt_bf16(float f) {
  return (unsigned short)((f32bits(f) + 0x8000u) >> 16);
}
__device__ __forceinline__ unsigned short cvt_bf16(unsigned short x) { return x; }

// pack {bf16(hi), bf16(lo)} in one v_perm_b32: result bytes [1:0] = src1[3:2],
// bytes [3:2] = src0[3:2]  (perm indices 0-3 pick src1, 4-7 pick src0)
__device__ __forceinline__ unsigned pack2_bf16(float lo, float hi) {
  return __builtin_amdgcn_perm(f32bits(hi) + 0x8000u, f32bits(lo) + 0x8000u,
                               0x07060302u);
}

// fp32 -> bf16 convert + store 16 elements to LDS (synchronous path)
__device__ __forceinline__ void stage_row16(const float* __restrict__ src,
                                            unsigned short* dst) {
  const float4* s4 = reinterpret_cast<const float4*>(src);
  unsigned o[8];
#pragma unroll
  for (int i = 0; i < 4; ++i) {
    float4 f = s4[i];
    o[i * 2 + 0] = pack2_bf16(f.x, f.y);
    o[i * 2 + 1] = pack2_bf16(f.z, f.w);
  }
  uint4* d4 = reinterpret_cast<uint4*>(dst);
  d4[0] = make_uint4(o[0], o[1], o[2], o[3]);
  d4[1] = make_uint4(o[4], o[5], o[6], o[7]);
}

// bf16 -> LDS: async DMA, no VGPR round trip. 32B = 2 x b128 beats.
// IOFFSET is added to BOTH the LDS and global address (ISA 15.18.3), so the
// second beat reuses the same base registers with offset:16.
__device__ __forceinline__ void stage_row16_async(
    const unsigned short* __restrict__ src, unsigned short* dst) {
  // low 32 bits of a generic __shared__ pointer == LDS byte offset
  unsigned lds_off = (unsigned)(unsigned long long)dst;
  asm volatile("global_load_async_to_lds_b128 %0, %1, off"
               :: "v"(lds_off), "v"(src) : "memory");
  asm volatile("global_load_async_to_lds_b128 %0, %1, off offset:16"
               :: "v"(lds_off), "v"(src) : "memory");
}

__device__ __forceinline__ void wait_asynccnt0() {
  asm volatile("s_wait_asynccnt 0" ::: "memory");
}

union Frag16 { v16bf v; unsigned u[8]; };

// C[M,N] = act( A[M,K] * B + bias ), A row-major.
// B_TRANS=1: B given as Bt[N][K] row-major.  B_TRANS=0: B[K][N] row-major.
// Outputs: Cf fp32, Cb bf16, CbT bf16 transposed per 2^tShift-row batch.
template <typename TA, typename TB, bool B_TRANS, bool HAS_BIAS, bool TANH_ACT,
          bool OUT_F32, bool OUT_BF16, bool OUT_T>
__global__ __launch_bounds__(256) void gemm_wmma_kernel(
    const TA* __restrict__ A, const TB* __restrict__ B,
    const float* __restrict__ bias,
    float* __restrict__ Cf, unsigned short* __restrict__ Cb,
    unsigned short* __restrict__ CbT,
    int M, int N, int K,
    long long zStrideA, long long zStrideB, long long zStrideC, int tShift) {
  using namespace cfg;
  constexpr bool ASYNC_A = IsBf16<TA>::value;
  constexpr bool ASYNC_B = B_TRANS && IsBf16<TB>::value;
  constexpr bool ANY_ASYNC = ASYNC_A || ASYNC_B;

  __shared__ unsigned short lds_a[2 * TILE];
  __shared__ unsigned short lds_b[2 * TILE];

  const int tid = threadIdx.x;
  const int z   = blockIdx.z;
  const int m0  = blockIdx.y * BM;
  const int n0  = blockIdx.x * BN;
  A += (long long)z * zStrideA;
  B += (long long)z * zStrideB;
  const long long zC = (long long)z * zStrideC;

  const int lane     = tid & 31;
  const int wave     = tid >> 5;
  const int wm       = wave >> 1;   // 0..3 -> 32-row strip
  const int wn       = wave & 1;    // 0..1 -> 64-col strip
  const int laneHalf = lane >> 4;   // 0/1
  const int lane15   = lane & 15;

  // per-thread staging coordinates
  const int srow  = tid >> 1;          // 0..127
  const int scolb = (tid & 1) * 16;    // 0 / 16
  const int krow  = tid >> 3;          // 0..31   (B non-transposed path)
  const int ncolb = (tid & 7) * 16;    // 0..112

  v8f acc[2][4];
#pragma unroll
  for (int mi = 0; mi < 2; ++mi)
#pragma unroll
    for (int ni = 0; ni < 4; ++ni)
      acc[mi][ni] = (v8f){0.f, 0.f, 0.f, 0.f, 0.f, 0.f, 0.f, 0.f};

  // ---- tile staging helpers ----
  auto stageA = [&](unsigned short* buf, int k0) {
    const TA* src = A + (long long)(m0 + srow) * K + k0 + scolb;
    unsigned short* dst = &buf[srow * LDSS + scolb];
    if constexpr (ASYNC_A) stage_row16_async(src, dst);
    else                   stage_row16(src, dst);
  };
  auto stageB = [&](unsigned short* buf, int k0) {
    if constexpr (B_TRANS) {
      const TB* src = B + (long long)(n0 + srow) * K + k0 + scolb;
      unsigned short* dst = &buf[srow * LDSS + scolb];
      if constexpr (ASYNC_B) stage_row16_async(src, dst);
      else                   stage_row16(src, dst);
    } else {
      const TB* src = B + (long long)(k0 + krow) * N + n0 + ncolb;
#pragma unroll
      for (int i = 0; i < 16; ++i)
        buf[(ncolb + i) * LDSS + krow] = cvt_bf16(src[i]);
    }
  };

  // ---- prologue: stage tile 0 into buffer 0 ----
  stageA(lds_a, 0);
  stageB(lds_b, 0);
  if constexpr (ANY_ASYNC) wait_asynccnt0();
  __syncthreads();

  int buf = 0;
#pragma unroll 2
  for (int k0 = 0; k0 < K; k0 += BK, buf ^= 1) {
    unsigned short* la = lds_a + buf * TILE;
    unsigned short* lb = lds_b + buf * TILE;

    // stage next tile into the other buffer while this one computes
    if (k0 + BK < K) {
      stageA(lds_a + (buf ^ 1) * TILE, k0 + BK);
      stageB(lds_b + (buf ^ 1) * TILE, k0 + BK);
    }

    // ---- build fragments (ISA 7.12.2 layouts) ----
    Frag16 fa[2], fb[4];
#pragma unroll
    for (int mi = 0; mi < 2; ++mi) {
      const int row = wm * 32 + mi * 16 + lane15;
#pragma unroll
      for (int j = 0; j < 8; ++j) {
        const int kk = ((j >> 2) << 4) + (laneHalf << 3) + ((j & 3) << 1);
        fa[mi].u[j] = *reinterpret_cast<const unsigned*>(&la[row * LDSS + kk]);
      }
    }
#pragma unroll
    for (int ni = 0; ni < 4; ++ni) {
      const int nrow = wn * 64 + ni * 16 + lane15;
#pragma unroll
      for (int j = 0; j < 8; ++j) {
        const int kk = ((j >> 2) << 4) + (laneHalf << 3) + ((j & 3) << 1);
        fb[ni].u[j] = *reinterpret_cast<const unsigned*>(&lb[nrow * LDSS + kk]);
      }
    }

#pragma unroll
    for (int mi = 0; mi < 2; ++mi)
#pragma unroll
      for (int ni = 0; ni < 4; ++ni)
        acc[mi][ni] = __builtin_amdgcn_wmma_f32_16x16x32_bf16(
            false, fa[mi].v, false, fb[ni].v, (short)0, acc[mi][ni], false,
            false);

    if constexpr (ANY_ASYNC) wait_asynccnt0();
    __syncthreads();
  }

  // ---- epilogue ----
#pragma unroll
  for (int ni = 0; ni < 4; ++ni) {
    const int gn = n0 + wn * 64 + ni * 16 + lane15;
    float bv = 0.f;
    if constexpr (HAS_BIAS) bv = bias[gn];
#pragma unroll
    for (int mi = 0; mi < 2; ++mi) {
      const int gmBase = m0 + wm * 32 + mi * 16 + laneHalf * 8;  // r = 0 row
      float vv[8];
#pragma unroll
      for (int r = 0; r < 8; ++r) {
        float v = acc[mi][ni][r];
        if constexpr (HAS_BIAS) v += bv;
        if constexpr (TANH_ACT) v = tanhf(v);
        vv[r] = v;
      }
      if constexpr (OUT_F32) {
#pragma unroll
        for (int r = 0; r < 8; ++r)
          Cf[zC + (long long)(gmBase + r) * N + gn] = vv[r];
      }
      if constexpr (OUT_BF16) {
#pragma unroll
        for (int r = 0; r < 8; ++r)
          Cb[zC + (long long)(gmBase + r) * N + gn] = cvt_bf16(vv[r]);
      }
      if constexpr (OUT_T) {
        // r-consecutive rows are contiguous in the transposed layout:
        // one 16B store of 8 packed bf16 per fragment per lane.
        const int bb = gmBase >> tShift;                 // batch index
        const int kr = gmBase & ((1 << tShift) - 1);     // multiple of 8
        const long long idx = (((long long)bb * N + gn) << tShift) + kr;
        uint4 o = make_uint4(pack2_bf16(vv[0], vv[1]), pack2_bf16(vv[2], vv[3]),
                             pack2_bf16(vv[4], vv[5]), pack2_bf16(vv[6], vv[7]));
        *reinterpret_cast<uint4*>(&CbT[idx]) = o;        // 16B aligned
      }
    }
  }
}

// Masked softmax over KLEN. tanh-bounded scores => no max-subtraction needed.
// softmax->mask->renorm == masked softmax (global denominator cancels).
// 8 contiguous elements per thread: float4 loads/stores, packed bf16 store.
__global__ __launch_bounds__(256) void softmax_mask_kernel(
    const float* __restrict__ score,      // [NB*QL][KL]
    const int* __restrict__ memlen,       // [NB]
    float* __restrict__ attnF,            // [NB*QL][KL] (into d_out)
    unsigned short* __restrict__ attnB) { // bf16 copy for next GEMM
  using namespace cfg;
  const int row = blockIdx.x;             // b*QL + q
  const int b   = row >> 9;               // QL == 512
  const int ml  = memlen[b];
  const int tid = threadIdx.x;
  const int base = tid * 8;               // 8 contiguous elems per thread

  const float4* s4 =
      reinterpret_cast<const float4*>(score + (long long)row * KL + base);
  float4 f0 = s4[0], f1 = s4[1];
  float e[8] = {f0.x, f0.y, f0.z, f0.w, f1.x, f1.y, f1.z, f1.w};
  float sum = 0.f;
#pragma unroll
  for (int i = 0; i < 8; ++i) {
    const float v = (base + i < ml) ? expf(e[i]) : 0.f;
    e[i] = v;
    sum += v;
  }
#pragma unroll
  for (int off = 16; off > 0; off >>= 1) sum += __shfl_xor(sum, off);
  __shared__ float red[8];
  if ((tid & 31) == 0) red[tid >> 5] = sum;
  __syncthreads();
  float total = 0.f;
#pragma unroll
  for (int w = 0; w < 8; ++w) total += red[w];
  const float inv = 1.0f / total;   // memory_len >= 1 -> total > 0

#pragma unroll
  for (int i = 0; i < 8; ++i) e[i] *= inv;

  float4* oF = reinterpret_cast<float4*>(attnF + (long long)row * KL + base);
  oF[0] = make_float4(e[0], e[1], e[2], e[3]);
  oF[1] = make_float4(e[4], e[5], e[6], e[7]);
  uint4 ob = make_uint4(pack2_bf16(e[0], e[1]), pack2_bf16(e[2], e[3]),
                        pack2_bf16(e[4], e[5]), pack2_bf16(e[6], e[7]));
  *reinterpret_cast<uint4*>(attnB + (long long)row * KL + base) = ob;
}

extern "C" void kernel_launch(void* const* d_in, const int* in_sizes, int n_in,
                              void* d_out, int out_size, void* d_ws,
                              size_t ws_size, hipStream_t stream) {
  using namespace cfg;
  (void)in_sizes; (void)n_in; (void)out_size; (void)ws_size;

  const float* k      = (const float*)d_in[0];
  const float* q      = (const float*)d_in[1];
  const int*   memlen = (const int*)d_in[2];
  const float* Wk     = (const float*)d_in[3];
  const float* bk     = (const float*)d_in[4];
  const float* Wq     = (const float*)d_in[5];
  const float* bq     = (const float*)d_in[6];
  const float* Wp     = (const float*)d_in[7];
  const float* bp     = (const float*)d_in[8];
  const float* weight = (const float*)d_in[9];

  float* out   = (float*)d_out;                          // [NB*QL][EMB]
  float* attnF = out + (size_t)NB * QL * HID;            // [NB*QL][KL]

  // workspace layout (256 MiB total)
  char* ws = (char*)d_ws;
  unsigned short* kx_bf   = (unsigned short*)(ws);                       // 64 MiB
  unsigned short* kxT_bf  = (unsigned short*)(ws + (67108864LL));        // 64 MiB
  unsigned short* qx_bf   = (unsigned short*)(ws + (134217728LL));       // 16 MiB (reused as mid)
  unsigned short* qw_bf   = (unsigned short*)(ws + (150994944LL));       // 16 MiB
  unsigned short* attn_bf = (unsigned short*)(ws + (167772160LL));       // 32 MiB
  float*          score_f = (float*)(ws + (201326592LL));                // 64 MiB
  unsigned short* mid_bf  = qx_bf;  // qx dead after step 3

  const dim3 blk(256);

  // 1) kx = k * Wk^T + bk  -> kx_bf [NB*KL][HID], kxT_bf [NB][HID][KL]
  gemm_wmma_kernel<float, float, true, true, false, false, true, true>
      <<<dim3(HID / BN, (NB * KL) / BM, 1), blk, 0, stream>>>(
          k, Wk, bk, nullptr, kx_bf, kxT_bf, NB * KL, HID, EMB, 0, 0, 0, 11);

  // 2) qx = q * Wq^T + bq  -> qx_bf [NB*QL][HID]
  gemm_wmma_kernel<float, float, true, true, false, false, true, false>
      <<<dim3(HID / BN, (NB * QL) / BM, 1), blk, 0, stream>>>(
          q, Wq, bq, nullptr, qx_bf, nullptr, NB * QL, HID, EMB, 0, 0, 0, 0);

  // 3) qw = qx * weight    -> qw_bf [NB*QL][HID]   (B not transposed)
  gemm_wmma_kernel<unsigned short, float, false, false, false, false, true, false>
      <<<dim3(HID / BN, (NB * QL) / BM, 1), blk, 0, stream>>>(
          qx_bf, weight, nullptr, nullptr, qw_bf, nullptr, NB * QL, HID, HID,
          0, 0, 0, 0);

  // 4) score = tanh(qw * kx^T) per batch -> score_f [NB][QL][KL]
  gemm_wmma_kernel<unsigned short, unsigned short, true, false, true, true, false, false>
      <<<dim3(KL / BN, QL / BM, NB), blk, 0, stream>>>(
          qw_bf, kx_bf, nullptr, score_f, nullptr, nullptr, QL, KL, HID,
          (long long)QL * HID, (long long)KL * HID, (long long)QL * KL, 0);

  // 5) masked softmax + renorm -> attnF (d_out), attn_bf
  softmax_mask_kernel<<<NB * QL, blk, 0, stream>>>(score_f, memlen, attnF,
                                                   attn_bf);

  // 6) mid = attn * kx per batch -> mid_bf [NB*QL][HID]
  gemm_wmma_kernel<unsigned short, unsigned short, true, false, false, false, true, false>
      <<<dim3(HID / BN, QL / BM, NB), blk, 0, stream>>>(
          attn_bf, kxT_bf, nullptr, nullptr, mid_bf, nullptr, QL, HID, KL,
          (long long)QL * KL, (long long)HID * KL, (long long)QL * HID, 0);

  // 7) out = mid * Wp^T + bp -> d_out fp32
  gemm_wmma_kernel<unsigned short, float, true, true, false, true, false, false>
      <<<dim3(EMB / BN, (NB * QL) / BM, 1), blk, 0, stream>>>(
          mid_bf, Wp, bp, out, nullptr, nullptr, NB * QL, EMB, HID, 0, 0, 0, 0);
}
